// RoPESwinTransformer_80564996538757
// MI455X (gfx1250) — matrix operand order repaired and checked by reading, not compiled, not run on hardware
//
#include <hip/hip_runtime.h>
#include <hip/hip_bf16.h>

typedef __attribute__((ext_vector_type(16))) _Float16 v16h;
typedef __attribute__((ext_vector_type(8)))  _Float16 v8h;
typedef __attribute__((ext_vector_type(8)))  float    v8f;

// exact vector type from the hipcc diagnostic for the async-LDS builtin
typedef int v4i_vs __attribute__((vector_size(4 * sizeof(int))));
typedef __attribute__((address_space(1))) v4i_vs* as1_v4i_p;
typedef __attribute__((address_space(3))) v4i_vs* as3_v4i_p;

#define WINSZ 7
#define NTOK  49
#define DIM   192
#define HEADS 6
#define HD    32
#define NWIN  64
#define BWIN  4096
#define NPAD  64

#define QKV_W_ELEMS (3 * DIM * DIM)   // 110592
#define PROJ_W_ELEMS (DIM * DIM)      // 36864
#define ATTN_WS_HALVES ((size_t)BWIN * NTOK * DIM)  // 38,535,168

#if __has_builtin(__builtin_amdgcn_global_load_async_to_lds_b128)
#define HAS_ASYNC_LDS 1
#else
#define HAS_ASYNC_LDS 0
#endif

__device__ inline void wait_async0() {
#if __has_builtin(__builtin_amdgcn_s_wait_asynccnt)
    __builtin_amdgcn_s_wait_asynccnt(0);
#else
    asm volatile("s_wait_asynccnt 0" ::: "memory");
#endif
}

// ---- fragment loaders -------------------------------------------------------
// A-matrix 16x32 f16 layout: lane L holds row (L%16);
// halves 0..15 -> k = kbase + 16*(L>=16) + i
__device__ inline v16h frag_a(const _Float16* base, int row0, int stride, int kbase) {
    const int lane = threadIdx.x & 31;
    const int lo = lane & 15, hi = lane >> 4;
    const _Float16* p = base + (size_t)(row0 + lo) * stride + kbase + hi * 16;
    v8h x0 = *(const v8h*)p;
    v8h x1 = *(const v8h*)(p + 8);
    return __builtin_shufflevector(x0, x1, 0,1,2,3,4,5,6,7,8,9,10,11,12,13,14,15);
}

// B-matrix 32x16 f16 (row-striped): lane L holds column (L%16);
// halves 0..15 -> k = kbase + 16*(L>=16) + i. Source stored as [col][k] row-major.
__device__ inline v16h frag_b(const _Float16* base, int col0, int stride, int kbase) {
    const int lane = threadIdx.x & 31;
    const int lo = lane & 15, hi = lane >> 4;
    const _Float16* p = base + (size_t)(col0 + lo) * stride + kbase + hi * 16;
    v8h x0 = *(const v8h*)p;
    v8h x1 = *(const v8h*)(p + 8);
    return __builtin_shufflevector(x0, x1, 0,1,2,3,4,5,6,7,8,9,10,11,12,13,14,15);
}

__device__ inline v8f zero8() {
    v8f z = {0.f, 0.f, 0.f, 0.f, 0.f, 0.f, 0.f, 0.f};
    return z;
}

// ---- kernel 0: one-shot fp32 -> f16 weight conversion -----------------------
__global__ __launch_bounds__(256)
void cvt_weights_kernel(const float* __restrict__ qkv_w, const float* __restrict__ proj_w,
                        _Float16* __restrict__ wh) {
    const int idx = blockIdx.x * 256 + threadIdx.x;
    if (idx < QKV_W_ELEMS) {
        wh[idx] = (_Float16)qkv_w[idx];
    } else if (idx < QKV_W_ELEMS + PROJ_W_ELEMS) {
        wh[idx] = (_Float16)proj_w[idx - QKV_W_ELEMS];
    }
}

// ---- kernel 1: fused QKV projection + RoPE + windowed attention -------------
__global__ __launch_bounds__(128)
void qkv_attn_kernel(const float* __restrict__ x, const float* __restrict__ mask,
                     const _Float16* __restrict__ wq_h, const float* __restrict__ qkv_b,
                     const float* __restrict__ rope, _Float16* __restrict__ attn_out) {
    __shared__ __align__(16) _Float16 lds_x[NPAD * DIM];   // x window, f16, zero-padded rows
    __shared__ __align__(16) _Float16 lds_q[NPAD * HD];    // q [token][chan]
    __shared__ __align__(16) _Float16 lds_k[NPAD * HD];    // k [token][chan]
    __shared__ __align__(16) _Float16 lds_vT[HD * NPAD];   // v transposed [chan][token]
    __shared__ __align__(16) _Float16 lds_p[NPAD * NPAD];  // softmax probs [qtok][ktok]

    const int b    = blockIdx.x / HEADS;
    const int h    = blockIdx.x % HEADS;
    const int tid  = threadIdx.x;
    const int w    = tid >> 5;         // wave id 0..3
    const int lane = tid & 31;
    const int lo   = lane & 15;
    const int hif  = lane >> 4;
    const int m0   = w * 16;           // this wave's M-tile base row

    // ---- stage x window into LDS (f16, rows 49..63 zeroed), 8 halves/iter ----
    for (int idx = tid; idx < (NPAD * DIM) / 8; idx += 128) {
        const int t = idx / (DIM / 8);
        const int c0 = (idx - t * (DIM / 8)) * 8;
        v8h hh;
        if (t < NTOK) {
            const float* sp = x + ((size_t)b * NTOK + t) * DIM + c0;
#pragma unroll
            for (int i = 0; i < 8; i++) hh[i] = (_Float16)sp[i];
        } else {
#pragma unroll
            for (int i = 0; i < 8; i++) hh[i] = (_Float16)0.f;
        }
        *(v8h*)&lds_x[t * DIM + c0] = hh;
    }
    __syncthreads();

    // ---- QKV GEMM for this head: 6 tiles of 16 channels, K=192 (6 steps) ----
    v16h afr[6];
#pragma unroll
    for (int ks = 0; ks < 6; ks++) afr[ks] = frag_a(lds_x, m0, DIM, ks * 32);

#pragma unroll
    for (int j = 0; j < 6; j++) {
        const int sel = j >> 1, part = j & 1;          // 0=q 1=k 2=v
        const int cb = sel * DIM + h * HD + part * 16; // global channel base
        v8f acc = zero8();
#pragma unroll
        for (int ks = 0; ks < 6; ks++) {
            v16h bfr = frag_b(wq_h, cb, DIM, ks * 32);
            acc = __builtin_amdgcn_wmma_f32_16x16x32_f16(
                false, afr[ks], false, bfr, (short)0, acc, false, false);
        }
        const float bias = qkv_b[cb + lo];
#pragma unroll
        for (int r = 0; r < 8; r++) {
            const int m = m0 + r + 8 * hif;
            const int nloc = part * 16 + lo;
            const _Float16 val = (_Float16)(acc[r] + bias);
            if (sel == 0)      lds_q[m * HD + nloc] = val;
            else if (sel == 1) lds_k[m * HD + nloc] = val;
            else               lds_vT[nloc * NPAD + m] = val;  // transpose v
        }
    }
    __syncthreads();

    // ---- RoPE on q (with 1/sqrt(hd) scale) and k, in LDS ----
    const float scale = 0.17677669529663687f; // 32^-0.5
#pragma unroll 1
    for (int idx = tid; idx < NTOK * (HD / 2); idx += 128) {
        const int t = idx >> 4, d = idx & 15;
        const float fx = rope[(0 * HEADS + h) * (HD / 2) + d];
        const float fy = rope[(1 * HEADS + h) * (HD / 2) + d];
        const float ang = (float)(t % WINSZ) * fx + (float)(t / WINSZ) * fy;
        float sn, cs;
        __sincosf(ang, &sn, &cs);
        float e = (float)lds_q[t * HD + 2 * d] * scale;
        float o = (float)lds_q[t * HD + 2 * d + 1] * scale;
        lds_q[t * HD + 2 * d]     = (_Float16)(e * cs - o * sn);
        lds_q[t * HD + 2 * d + 1] = (_Float16)(e * sn + o * cs);
        e = (float)lds_k[t * HD + 2 * d];
        o = (float)lds_k[t * HD + 2 * d + 1];
        lds_k[t * HD + 2 * d]     = (_Float16)(e * cs - o * sn);
        lds_k[t * HD + 2 * d + 1] = (_Float16)(e * sn + o * cs);
    }
    __syncthreads();

    // ---- S = q @ k^T  (K = hd = 32 -> exactly one WMMA per tile) ----
    v8f s[4];
    {
        const v16h aq = frag_a(lds_q, m0, HD, 0);
#pragma unroll
        for (int nt = 0; nt < 4; nt++) {
            const v16h bk = frag_b(lds_k, nt * 16, HD, 0);
            s[nt] = __builtin_amdgcn_wmma_f32_16x16x32_f16(
                false, aq, false, bk, (short)0, zero8(), false, false);
        }
    }

    // ---- softmax rows (row m spread over 16 lanes sharing hif; xor<16 stays in half) ----
    const int widx = b % NWIN;
    const float* mbase = mask + (size_t)widx * NTOK * NTOK;
#pragma unroll
    for (int r = 0; r < 8; r++) {
        const int m = m0 + r + 8 * hif;
        float vals[4];
#pragma unroll
        for (int nt = 0; nt < 4; nt++) {
            const int n = nt * 16 + lo;
            float sv = s[nt][r];
            if (n < NTOK) {
                if (m < NTOK) sv += mbase[m * NTOK + n];
            } else {
                sv = -1e30f;
            }
            vals[nt] = sv;
        }
        float mx = fmaxf(fmaxf(vals[0], vals[1]), fmaxf(vals[2], vals[3]));
#pragma unroll
        for (int d = 8; d >= 1; d >>= 1) mx = fmaxf(mx, __shfl_xor(mx, d, 32));
        float sum = 0.f;
#pragma unroll
        for (int nt = 0; nt < 4; nt++) { vals[nt] = __expf(vals[nt] - mx); sum += vals[nt]; }
#pragma unroll
        for (int d = 8; d >= 1; d >>= 1) sum += __shfl_xor(sum, d, 32);
        const float inv = 1.f / sum;
#pragma unroll
        for (int nt = 0; nt < 4; nt++)
            lds_p[m * NPAD + nt * 16 + lo] = (_Float16)(vals[nt] * inv);
    }
    __syncthreads();

    // ---- out = P @ V  (K = 64 -> 2 steps, 2 channel tiles) ----
    v8f o0 = zero8(), o1 = zero8();
#pragma unroll
    for (int kb = 0; kb < 2; kb++) {
        const v16h ap = frag_a(lds_p, m0, NPAD, kb * 32);
        const v16h bv0 = frag_b(lds_vT, 0, NPAD, kb * 32);
        const v16h bv1 = frag_b(lds_vT, 16, NPAD, kb * 32);
        o0 = __builtin_amdgcn_wmma_f32_16x16x32_f16(false, ap, false, bv0, (short)0, o0, false, false);
        o1 = __builtin_amdgcn_wmma_f32_16x16x32_f16(false, ap, false, bv1, (short)0, o1, false, false);
    }
#pragma unroll
    for (int r = 0; r < 8; r++) {
        const int m = m0 + r + 8 * hif;
        if (m < NTOK) {
            _Float16* dst = attn_out + ((size_t)b * NTOK + m) * DIM + h * HD;
            dst[lo]      = (_Float16)o0[r];
            dst[16 + lo] = (_Float16)o1[r];
        }
    }
}

// ---- kernel 2: output projection [200704,192]f16 x proj_w^T + bias -> fp32 --
// Per-wave A-tile (16x192 f16 = 6144 contiguous bytes) staged to LDS via
// async global->LDS b128 transfers (gfx1250 path), then consumed as fragments.
__global__ __launch_bounds__(128)
void proj_kernel(const _Float16* __restrict__ a, const _Float16* __restrict__ wp_h,
                 const float* __restrict__ pb, float* __restrict__ out) {
    __shared__ __align__(16) _Float16 lds_a[4 * 16 * DIM];  // 24 KB, one tile per wave

    const int lane = threadIdx.x & 31;
    const int lo = lane & 15, hif = lane >> 4;
    const int w = threadIdx.x >> 5;
    const int mtile = blockIdx.x * 4 + w;
    const int m0 = mtile * 16;

    _Float16* wbuf = lds_a + w * 16 * DIM;
    const _Float16* gbase = a + (size_t)m0 * DIM;

    // stage the contiguous 6144-byte A-tile: 12 x (32 lanes x 16B)
#if HAS_ASYNC_LDS
#pragma unroll
    for (int c = 0; c < 12; c++) {
        const int off = (c * 32 + lane) * 8;  // halves
        __builtin_amdgcn_global_load_async_to_lds_b128(
            (as1_v4i_p)(gbase + off),
            (as3_v4i_p)(wbuf + off), 0, 0);
    }
    wait_async0();
#else
#pragma unroll
    for (int c = 0; c < 12; c++) {
        const int off = (c * 32 + lane) * 8;
        *(v8h*)(wbuf + off) = *(const v8h*)(gbase + off);
    }
#endif

    v16h afr[6];
#pragma unroll
    for (int ks = 0; ks < 6; ks++) afr[ks] = frag_a(wbuf, 0, DIM, ks * 32);

#pragma unroll
    for (int nt = 0; nt < 12; nt++) {
        const float bias = pb[nt * 16 + lo];
        v8f acc;
#pragma unroll
        for (int r = 0; r < 8; r++) acc[r] = bias;
#pragma unroll
        for (int ks = 0; ks < 6; ks++) {
            v16h bfr = frag_b(wp_h, nt * 16, DIM, ks * 32);
            acc = __builtin_amdgcn_wmma_f32_16x16x32_f16(
                false, afr[ks], false, bfr, (short)0, acc, false, false);
        }
#pragma unroll
        for (int r = 0; r < 8; r++) {
            const size_t m = (size_t)m0 + r + 8 * hif;
            out[m * DIM + nt * 16 + lo] = acc[r];
        }
    }
}

extern "C" void kernel_launch(void* const* d_in, const int* in_sizes, int n_in,
                              void* d_out, int out_size, void* d_ws, size_t ws_size,
                              hipStream_t stream) {
    const float* x      = (const float*)d_in[0];
    const float* mask   = (const float*)d_in[1];
    const float* qkv_w  = (const float*)d_in[2];
    const float* qkv_b  = (const float*)d_in[3];
    const float* proj_w = (const float*)d_in[4];
    const float* proj_b = (const float*)d_in[5];
    const float* rope   = (const float*)d_in[6];
    float* out = (float*)d_out;

    // workspace layout: [attn f16 | qkv_w f16 | proj_w f16]
    _Float16* attn_ws = (_Float16*)d_ws;
    _Float16* wq_h = attn_ws + ATTN_WS_HALVES;
    _Float16* wp_h = wq_h + QKV_W_ELEMS;

    const int cvt_total = QKV_W_ELEMS + PROJ_W_ELEMS;
    cvt_weights_kernel<<<(cvt_total + 255) / 256, 256, 0, stream>>>(qkv_w, proj_w, wq_h);

    qkv_attn_kernel<<<BWIN * HEADS, 128, 0, stream>>>(x, mask, wq_h, qkv_b, rope, attn_ws);

    const int mtiles = (BWIN * NTOK) / 16;   // 12544
    proj_kernel<<<mtiles / 4, 128, 0, stream>>>(attn_ws, wp_h, proj_b, out);
}
